// ElectronicEmbedding_32899449487751
// MI455X (gfx1250) — matrix operand order repaired and checked by reading, not compiled, not run on hardware
//
#include <hip/hip_runtime.h>
#include <math.h>

// ---------------------------------------------------------------------------
// Fused ElectronicEmbedding for MI455X (gfx1250, wave32, WMMA bf16 + TDM).
// One workgroup (256 threads = 8 wave32) per molecule (64 atoms x 256 feats).
// GEMM1 is algebraically folded into a mat-vec (W_lin @ k precomputed fp32).
// GEMMs 2-4 run on v_wmma_f32_16x16x32_bf16 with fp32 accumulation.
// Weight K-slabs are streamed into double-buffered LDS by the Tensor Data
// Mover (tensor_load_to_lds + s_wait_tensorcnt); waves only ds_load + wmma.
// ---------------------------------------------------------------------------

#define F_DIM  256
#define NATOMS 64
#define LDA    264                 // bf16 activation row stride (pad, 16B-aligned rows)
#define LDF    260                 // f32 residual row stride (pad, 16B-aligned rows)
#define LDW    40                  // bf16 weight-slab row stride: 32 K-vals + 16B pad
#define SLAB   (F_DIM * LDW)       // elems per weight slab buffer

typedef __bf16 bf16_t;
typedef __attribute__((ext_vector_type(16))) __bf16 v16bf;
typedef __attribute__((ext_vector_type(8)))  __bf16 v8bf;
typedef __attribute__((ext_vector_type(4)))  __bf16 v4bf;
typedef __attribute__((ext_vector_type(8)))  float  v8f;
typedef __attribute__((ext_vector_type(4)))  unsigned int v4u;
typedef __attribute__((ext_vector_type(8)))  int v8i;
typedef __attribute__((ext_vector_type(4)))  int v4i;

union V16 { v16bf v; struct { v8bf lo, hi; } p; };

#if __has_builtin(__builtin_amdgcn_tensor_load_to_lds) && \
    __has_builtin(__builtin_amdgcn_s_wait_tensorcnt)
#define USE_TDM 1
#else
#define USE_TDM 0
#endif

__device__ __forceinline__ bf16_t f2bf(float f) {
  unsigned int u = __builtin_bit_cast(unsigned int, f);
  unsigned int r = u + 0x7FFFu + ((u >> 16) & 1u);   // round-to-nearest-even
  return __builtin_bit_cast(bf16_t, (unsigned short)(r >> 16));
}
__device__ __forceinline__ float sigm(float x) { return 1.0f / (1.0f + __expf(-x)); }

__device__ __forceinline__ void zero_acc(v8f acc[8]) {
#pragma unroll
  for (int j = 0; j < 8; ++j) {
    v8f z = {0.f, 0.f, 0.f, 0.f, 0.f, 0.f, 0.f, 0.f};
    acc[j] = z;
  }
}

#if USE_TDM
typedef __attribute__((address_space(3))) char lds_char_t;
__device__ __forceinline__ unsigned int lds_byte_off(const void* p) {
  // generic -> LDS addrspacecast; ptrtoint gives the workgroup-relative offset
  return (unsigned int)(unsigned long long)(lds_char_t*)(void*)(const_cast<void*>(p));
}

// Issue one TDM descriptor: copy a 32(K) x 256(f) bf16 tile (tile start gsrc,
// row stride 256 elems) into LDS at lds_byte, inserting 16B of pad after every
// 64B row => LDS row stride 80B (= LDW bf16 elems).  ISA 8.3/8.4 packing.
__device__ __forceinline__ void tdm_load_slab(const bf16_t* gsrc, unsigned int lds_byte) {
  unsigned long long G = (unsigned long long)gsrc;
  v4u g0;
  g0.x = 1u;                                           // count=1, load, user mode
  g0.y = lds_byte;                                     // lds_addr
  g0.z = (unsigned int)(G & 0xFFFFFFFFu);              // global_addr[31:0]
  g0.w = (unsigned int)((G >> 32) & 0x01FFFFFFu)       // global_addr[56:32]
       | (2u << 30);                                   // type=2 (image)
  v8i g1;
  g1[0] = (int)((1u << 16)      // data_size = 2 bytes
              | (1u << 20)      // pad_enable
              | (3u << 22)      // pad_interval: 16 DWORDs = 64B
              | (3u << 25));    // pad_amount:   4 DWORDs = 16B
  g1[1] = (int)(256u << 16);    // tensor_dim0[15:0] = 256   (atomic_barrier_addr=0)
  g1[2] = (int)(256u << 16);    // tensor_dim0 hi=0 ; tensor_dim1[15:0] = 256
  g1[3] = (int)(32u  << 16);    // tensor_dim1 hi=0 ; tile_dim0 = 32
  g1[4] = 256;                  // tile_dim1 = 256 ; tile_dim2 = 0
  g1[5] = 256;                  // tensor_dim0_stride[31:0] = 256
  g1[6] = (int)(256u << 16);    // stride0 hi=0 ; tensor_dim1_stride[15:0] = 256
  g1[7] = 0;                    // tensor_dim1_stride hi = 0
  v4i g2 = {0, 0, 0, 0};        // 2D tile: higher dims unused
  v4i g3 = {0, 0, 0, 0};
#if defined(__clang_major__) && (__clang_major__ >= 23)
  v8i g4 = {0, 0, 0, 0, 0, 0, 0, 0};   // clang-23 toolchain: 6-arg form
  __builtin_amdgcn_tensor_load_to_lds(g0, g1, g2, g3, g4, 0);
#else
  __builtin_amdgcn_tensor_load_to_lds(g0, g1, g2, g3, 0);      // ROCm 7.2: 5-arg
#endif
}
#endif

// ---------------------------------------------------------------------------
// Prep kernel 1: transpose+convert the three GEMM weights to bf16, f-major:
//   Wt[w][f][k] = bf16(W_w[k][f]),  w in {W_r1, W_r2, W_out}
// ---------------------------------------------------------------------------
__global__ void prep_weights(const float* __restrict__ Wr1,
                             const float* __restrict__ Wr2,
                             const float* __restrict__ Wout,
                             bf16_t* __restrict__ ws) {
  int idx = blockIdx.x * blockDim.x + threadIdx.x;   // 0 .. 3*65536-1
  int w   = idx >> 16;
  int rem = idx & 65535;
  int f   = rem >> 8;
  int k   = rem & 255;
  const float* src = (w == 0) ? Wr1 : (w == 1) ? Wr2 : Wout;
  ws[idx] = f2bf(src[k * F_DIM + f]);
}

// ---------------------------------------------------------------------------
// Prep kernel 2: fold the first linear into the attention dot:
//   wt[s][k] = sum_f W_lin[k][f] * k_s[f],   cb[s] = sum_f b_lin[f] * k_s[f]
// ---------------------------------------------------------------------------
__global__ void prep_kvec(const float* __restrict__ W_lin,
                          const float* __restrict__ b_lin,
                          const float* __restrict__ k_plus,
                          const float* __restrict__ k_minus,
                          float* __restrict__ wt) {
  const float* kv = (blockIdx.x == 0) ? k_plus : k_minus;
  int k = threadIdx.x;
  const float* row = W_lin + k * F_DIM;
  float s = 0.f;
  for (int f = 0; f < F_DIM; ++f) s += row[f] * kv[f];
  wt[blockIdx.x * F_DIM + k] = s;
  if (k == 0) {
    float c = 0.f;
    for (int f = 0; f < F_DIM; ++f) c += b_lin[f] * kv[f];
    wt[2 * F_DIM + blockIdx.x] = c;
  }
}

// ---------------------------------------------------------------------------
// Core 64x256 @ 256x256 tile GEMM on WMMA bf16.
// Wave w owns row-tile (w&3) and col-tiles [(w>>2)*8, +8).  K consumed in
// 32-wide slabs double-buffered in LDS.  TDM path: wave 0 issues one
// tensor_load_to_lds per slab; compute waves never touch global memory.
// ---------------------------------------------------------------------------
__device__ __forceinline__ void gemm_tile(const bf16_t* __restrict__ Wt,   // [256][256] f-major bf16 (global)
                                          const bf16_t* shA,               // [64][LDA] bf16 A tile (LDS)
                                          bf16_t* shW,                     // 2 slabs [256][LDW] (LDS)
                                          int tid, int half, int ln16,
                                          int row_tile, int ct_base,
                                          v8f acc[8]) {
#if USE_TDM
  const bool issuer = (tid < 32);                 // wave 0 drives the TDM
  if (issuer) {
    tdm_load_slab(Wt, lds_byte_off(shW));         // slab 0 -> buffer 0
    __builtin_amdgcn_s_wait_tensorcnt(0);
  }
  __syncthreads();                                // slab 0 visible to all waves
#else
  const bf16_t* gsrc = Wt + tid * F_DIM;          // this thread stages Wt row f=tid
  v8bf r0 = *(const v8bf*)(gsrc + 0);
  v8bf r1 = *(const v8bf*)(gsrc + 8);
  v8bf r2 = *(const v8bf*)(gsrc + 16);
  v8bf r3 = *(const v8bf*)(gsrc + 24);
  {
    bf16_t* w = shW + tid * LDW;
    *(v8bf*)(w + 0)  = r0;  *(v8bf*)(w + 8)  = r1;
    *(v8bf*)(w + 16) = r2;  *(v8bf*)(w + 24) = r3;
  }
  __syncthreads();
#endif

  const int arow = (row_tile * 16 + ln16) * LDA + 8 * half;

#pragma unroll
  for (int ks = 0; ks < 8; ++ks) {
    const int buf = ks & 1;

#if USE_TDM
    if (ks < 7 && issuer)                         // stream next slab via TDM
      tdm_load_slab(Wt + (ks + 1) * 32, lds_byte_off(shW + (buf ^ 1) * SLAB));
#else
    if (ks < 7) {                                 // pipeline next slab into regs
      const bf16_t* g = gsrc + (ks + 1) * 32;
      r0 = *(const v8bf*)(g + 0);   r1 = *(const v8bf*)(g + 8);
      r2 = *(const v8bf*)(g + 16);  r3 = *(const v8bf*)(g + 24);
      if (ks < 6) __builtin_prefetch(gsrc + (ks + 2) * 32, 0, 1);
    }
#endif

    // A fragment (16x32 bf16): lane holds row m = rt*16+ln16;
    // elems 0..7 -> K = ks*32 + 8*half + e ; elems 8..15 -> +16.
    V16 a;
    const bf16_t* pa = shA + arow + ks * 32;
    a.p.lo = *(const v8bf*)(pa);
    a.p.hi = *(const v8bf*)(pa + 16);

#pragma unroll
    for (int j = 0; j < 8; ++j) {
      // B fragment (32x16 bf16): lane holds col n; elems 0..7 -> K = 8*half + e,
      // elems 8..15 -> +16 within the slab.
      V16 b;
      const bf16_t* pb = shW + buf * SLAB + ((ct_base + j) * 16 + ln16) * LDW + 8 * half;
      b.p.lo = *(const v8bf*)(pb);
      b.p.hi = *(const v8bf*)(pb + 16);
      acc[j] = __builtin_amdgcn_wmma_f32_16x16x32_bf16(
          false, a.v, false, b.v, (short)0, acc[j], false, false);
    }

#if USE_TDM
    __syncthreads();                              // all reads of buf done
    if (ks < 7) {
      if (issuer) __builtin_amdgcn_s_wait_tensorcnt(0);   // slab ks+1 landed
      __syncthreads();                            // publish to all waves
    }
#else
    if (ks < 7) {                                 // commit next slab to other buffer
      bf16_t* w = shW + (buf ^ 1) * SLAB + tid * LDW;
      *(v8bf*)(w + 0)  = r0;  *(v8bf*)(w + 8)  = r1;
      *(v8bf*)(w + 16) = r2;  *(v8bf*)(w + 24) = r3;
    }
    __syncthreads();
#endif
  }
}

// ---------------------------------------------------------------------------
// Main fused kernel: one block per molecule.
// ---------------------------------------------------------------------------
__global__ __launch_bounds__(256)
void electronic_embedding_kernel(const float* __restrict__ psi,
                                 const float* __restrict__ e_z,
                                 const float* __restrict__ v_plus,
                                 const float* __restrict__ v_minus,
                                 const float* __restrict__ b_out,
                                 const float* __restrict__ a1, const float* __restrict__ be1,
                                 const float* __restrict__ a2, const float* __restrict__ be2,
                                 const float* __restrict__ a3, const float* __restrict__ be3,
                                 const bf16_t* __restrict__ Wt,   // 3 x [256][256] f-major bf16
                                 const float* __restrict__ wtk,   // [2][256] + cb[2]
                                 float* __restrict__ out) {
  extern __shared__ char smem[];
  float*  shF   = (float*)smem;                         // [64][LDF] fp32 (av residual)
  bf16_t* shW   = (bf16_t*)(shF + NATOMS * LDF);        // 2 * SLAB
  bf16_t* shA   = (bf16_t*)(shW + 2 * SLAB);            // [64][LDA] bf16 activations
  float*  shArg = (float*)(shA + NATOMS * LDA);         // [64]
  float*  shAi  = shArg + NATOMS;                       // [64]
  float*  shP   = shAi + NATOMS;                        // [256] dot partials

  const int tid  = threadIdx.x;
  const int lane = tid & 31;
  const int half = lane >> 4;
  const int ln16 = lane & 15;
  const int wave = tid >> 5;
  const int row_tile = wave & 3;
  const int ct_base  = (wave >> 2) * 8;
  const int mol = blockIdx.x;

  const float psim = psi[mol];
  const bool  pos  = (psim >= 0.0f);                    // uniform per block
  const float* wt   = pos ? wtk : (wtk + F_DIM);
  const float  cb   = pos ? wtk[2 * F_DIM] : wtk[2 * F_DIM + 1];
  const float* vvec = pos ? v_plus : v_minus;

  const float* ez = e_z + (size_t)mol * NATOMS * F_DIM;

  // ---- Phase A: arg[n] = (e_z[n,:] . (W_lin@k) + b.k) / sqrt(F), deterministic ----
  {
    const int atom = tid >> 2;
    const int q    = (tid & 3) * 64;
    const float* er = ez + atom * F_DIM + q;
    const float* wr = wt + q;
    float s = 0.f;
#pragma unroll
    for (int j = 0; j < 16; ++j) {
      float4 x  = *(const float4*)(er + j * 4);
      float4 w4 = *(const float4*)(wr + j * 4);
      s += x.x * w4.x + x.y * w4.y + x.z * w4.z + x.w * w4.w;
    }
    shP[tid] = s;
  }
  __syncthreads();
  if (tid < NATOMS) {
    float dot = shP[4 * tid] + shP[4 * tid + 1] + shP[4 * tid + 2] + shP[4 * tid + 3];
    float arg = (dot + cb) * 0.0625f;                   // / sqrt(256)
    shArg[tid] = (arg > 20.0f) ? arg : log1pf(__expf(arg));   // softplus
  }
  __syncthreads();
  if (tid < NATOMS) {
    float denom = 0.0f;
#pragma unroll
    for (int i = 0; i < NATOMS; ++i) denom += shArg[i]; // fixed order: deterministic
    shAi[tid] = psim * shArg[tid] / denom;              // a_i
  }
  __syncthreads();

  // ---- Phase B: av = a_i * v ; keep fp32 residual ; shA = bf16(swish1(av)) ----
#pragma unroll
  for (int g = 0; g < 16; ++g) {
    int idx  = tid * 4 + g * 1024;
    int atom = idx >> 8;
    int col  = idx & 255;
    float ai = shAi[atom];
    float4 vv = *(const float4*)(vvec + col);
    float4 al = *(const float4*)(a1 + col);
    float4 be = *(const float4*)(be1 + col);
    float av0 = ai * vv.x, av1 = ai * vv.y, av2 = ai * vv.z, av3 = ai * vv.w;
    *(float4*)(shF + atom * LDF + col) = make_float4(av0, av1, av2, av3);
    v4bf t;
    t[0] = f2bf(al.x * av0 * sigm(be.x * av0));
    t[1] = f2bf(al.y * av1 * sigm(be.y * av1));
    t[2] = f2bf(al.z * av2 * sigm(be.z * av2));
    t[3] = f2bf(al.w * av3 * sigm(be.w * av3));
    *(v4bf*)(shA + atom * LDA + col) = t;
  }
  __syncthreads();

  v8f acc[8];

  // ---- GEMM 2: u = swish1(av) @ W_r1 ; shA = bf16(swish2(u)) ----
  zero_acc(acc);
  gemm_tile(Wt + 0 * 65536, shA, shW, tid, half, ln16, row_tile, ct_base, acc);
#pragma unroll
  for (int j = 0; j < 8; ++j) {
    int fc = (ct_base + j) * 16 + ln16;
    float al = a2[fc], bb = be2[fc];
#pragma unroll
    for (int v = 0; v < 8; ++v) {
      float u = acc[j][v];
      shA[(row_tile * 16 + 8 * half + v) * LDA + fc] = f2bf(al * u * sigm(bb * u));
    }
  }
  __syncthreads();

  // ---- GEMM 3: h = av + (...) @ W_r2 ; shA = bf16(swish3(h)) ----
  zero_acc(acc);
  gemm_tile(Wt + 1 * 65536, shA, shW, tid, half, ln16, row_tile, ct_base, acc);
#pragma unroll
  for (int j = 0; j < 8; ++j) {
    int fc = (ct_base + j) * 16 + ln16;
    float al = a3[fc], bb = be3[fc];
#pragma unroll
    for (int v = 0; v < 8; ++v) {
      int atom = row_tile * 16 + 8 * half + v;
      float h = acc[j][v] + shF[atom * LDF + fc];       // residual add (fp32)
      shA[atom * LDA + fc] = f2bf(al * h * sigm(bb * h));
    }
  }
  __syncthreads();

  // ---- GEMM 4: e_psi = swish3(h) @ W_out + b_out -> global fp32 ----
  zero_acc(acc);
  gemm_tile(Wt + 2 * 65536, shA, shW, tid, half, ln16, row_tile, ct_base, acc);
  float* gout = out + (size_t)mol * NATOMS * F_DIM;
#pragma unroll
  for (int j = 0; j < 8; ++j) {
    int fc = (ct_base + j) * 16 + ln16;
    float bo = b_out[fc];
#pragma unroll
    for (int v = 0; v < 8; ++v) {
      int atom = row_tile * 16 + 8 * half + v;
      gout[atom * F_DIM + fc] = acc[j][v] + bo;
    }
  }
}

// ---------------------------------------------------------------------------
extern "C" void kernel_launch(void* const* d_in, const int* in_sizes, int n_in,
                              void* d_out, int out_size, void* d_ws, size_t ws_size,
                              hipStream_t stream) {
  (void)n_in; (void)out_size; (void)ws_size;
  const float* psi     = (const float*)d_in[0];
  const float* e_z     = (const float*)d_in[1];
  /* d_in[2] = num_atoms (constant 64 per setup) */
  const float* W_lin   = (const float*)d_in[3];
  const float* b_lin   = (const float*)d_in[4];
  const float* k_plus  = (const float*)d_in[5];
  const float* k_minus = (const float*)d_in[6];
  const float* v_plus  = (const float*)d_in[7];
  const float* v_minus = (const float*)d_in[8];
  const float* W_r1    = (const float*)d_in[9];
  const float* W_r2    = (const float*)d_in[10];
  const float* W_out   = (const float*)d_in[11];
  const float* b_out   = (const float*)d_in[12];
  const float* a1  = (const float*)d_in[13];
  const float* be1 = (const float*)d_in[14];
  const float* a2  = (const float*)d_in[15];
  const float* be2 = (const float*)d_in[16];
  const float* a3  = (const float*)d_in[17];
  const float* be3 = (const float*)d_in[18];
  float* out = (float*)d_out;

  const int n_mol = in_sizes[0];                 // 4096

  // Workspace layout: 3x bf16 [256][256] transposed weights, then fp32
  // wt[2][256] + cb[2].
  bf16_t* Wt  = (bf16_t*)d_ws;
  float*  wtk = (float*)((char*)d_ws + 3 * 65536 * sizeof(bf16_t));

  prep_weights<<<(3 * 65536) / 256, 256, 0, stream>>>(W_r1, W_r2, W_out, Wt);
  prep_kvec<<<2, 256, 0, stream>>>(W_lin, b_lin, k_plus, k_minus, wtk);

  size_t shmem = (size_t)(NATOMS * LDF) * sizeof(float)      // residual
               + (size_t)(2 * SLAB) * sizeof(bf16_t)         // weight slabs
               + (size_t)(NATOMS * LDA) * sizeof(bf16_t)     // activations
               + (size_t)(2 * NATOMS + 256) * sizeof(float); // arg/ai/partials
  (void)hipFuncSetAttribute((const void*)electronic_embedding_kernel,
                            hipFuncAttributeMaxDynamicSharedMemorySize, (int)shmem);

  electronic_embedding_kernel<<<n_mol, 256, shmem, stream>>>(
      psi, e_z, v_plus, v_minus, b_out,
      a1, be1, a2, be2, a3, be3, Wt, wtk, out);
}
